// Encoder_5617817223522
// MI455X (gfx1250) — matrix operand order repaired and checked by reading, not compiled, not run on hardware
//
#include <hip/hip_runtime.h>
#include <hip/hip_bf16.h>
#include <math.h>

#define T_DATA_C 50000
#define T_NO_C   200
#define SUB_NO_C 20
#define HID_C    500
#define WIN_C    399

#define K1P 416             // WIN padded to multiple of 32
#define NP  512             // HID padded
#define KCONV 224           // conv window padded to multiple of 32
#define NCONV 32            // conv/head N padded (20 -> 32)
#define BM  128
#define BN  128
#define WMT 64
#define WNT 32
#define MBLK 391            // ceil(50000/128)
#define MROWS (MBLK * BM)   // 50048
#define VPAD_LEN 50464      // covers (MROWS-1) + (K1P-1)  = 50462
#define VPAD_SYN 50272      // covers (MROWS-1) + (KCONV-1) = 50270

typedef __bf16 bf16;
typedef bf16 v16bf __attribute__((ext_vector_type(16)));
typedef bf16 v8bf  __attribute__((ext_vector_type(8)));
typedef float v8f  __attribute__((ext_vector_type(8)));

// ---------------- fragment loaders (wave32 WMMA layouts, ISA 7.12.2) --------

// A 16x32 bf16 row-major, aligned leading dim.
// lane<16: M=lane, K = {0..7, 16..23}; lane>=16: M=lane-16, K = {8..15, 24..31}
__device__ __forceinline__ v16bf frag_a_vec(const bf16* corner, int ld) {
  const int lane = threadIdx.x & 31;
  const int half = lane >> 4;
  const int r    = lane & 15;
  const bf16* p = corner + (size_t)r * ld + half * 8;
  v8bf lo = *(const v8bf*)(p);
  v8bf hi = *(const v8bf*)(p + 16);
  v16bf o;
#pragma unroll
  for (int i = 0; i < 8; ++i) { o[i] = lo[i]; o[i + 8] = hi[i]; }
  return o;
}

// A 16x32 bf16 where A[t][k] = vpad[t+k] (sliding window, lda==1).
__device__ __forceinline__ v16bf frag_a_win(const bf16* corner) {
  const int lane = threadIdx.x & 31;
  const int half = lane >> 4;
  const int r    = lane & 15;
  const bf16* p = corner + r + half * 8;
  v16bf o;
#pragma unroll
  for (int i = 0; i < 8; ++i) { o[i] = p[i]; o[i + 8] = p[16 + i]; }
  return o;
}

// B 32x16 bf16 where logical B[k][n] = W[n][k], W row-major [N,K].
// lane<16: N=lane, K = {0..7, 16..23}; lane>=16: N=lane-16, K = {8..15, 24..31}
__device__ __forceinline__ v16bf frag_b(const bf16* corner, int ld) {
  const int lane = threadIdx.x & 31;
  const int half = lane >> 4;
  const int n    = lane & 15;
  const bf16* p = corner + (size_t)n * ld + half * 8;
  v8bf lo = *(const v8bf*)(p);
  v8bf hi = *(const v8bf*)(p + 16);
  v16bf o;
#pragma unroll
  for (int i = 0; i < 8; ++i) { o[i] = lo[i]; o[i + 8] = hi[i]; }
  return o;
}

#define WMMA_BF16(a, b, c) \
  __builtin_amdgcn_wmma_f32_16x16x32_bf16(false, (a), false, (b), (short)0, (c), false, false)

// ---------------- hidden-layer WMMA GEMM: H = lrelu(A @ W^T + b) ------------
template <bool AWIN>
__global__ __launch_bounds__(256, 2) void mlp_gemm(
    const bf16* __restrict__ A, int lda,
    const bf16* __restrict__ W, int ldw,
    const float* __restrict__ bias, int K,
    bf16* __restrict__ Hout) {
  const int wid  = threadIdx.x >> 5;
  const int lane = threadIdx.x & 31;
  const int wm = wid >> 2, wn = wid & 3;          // 2x4 wave grid
  const int t0 = blockIdx.x * BM + wm * WMT;      // wave row base
  const int n0 = blockIdx.y * BN + wn * WNT;      // wave col base

  v8f acc[4][2];
#pragma unroll
  for (int i = 0; i < 4; ++i)
#pragma unroll
    for (int j = 0; j < 2; ++j)
#pragma unroll
      for (int e = 0; e < 8; ++e) acc[i][j][e] = 0.f;

  for (int k0 = 0; k0 < K; k0 += 32) {
    v16bf bfr[2];
#pragma unroll
    for (int nf = 0; nf < 2; ++nf)
      bfr[nf] = frag_b(W + (size_t)(n0 + nf * 16) * ldw + k0, ldw);
#pragma unroll
    for (int mf = 0; mf < 4; ++mf) {
      v16bf afr;
      if (AWIN) afr = frag_a_win(A + (t0 + mf * 16) + k0);
      else      afr = frag_a_vec(A + (size_t)(t0 + mf * 16) * lda + k0, lda);
#pragma unroll
      for (int nf = 0; nf < 2; ++nf)
        acc[mf][nf] = WMMA_BF16(afr, bfr[nf], acc[mf][nf]);
    }
    if (k0 + 32 < K)  // hint next W K-slice into cache (global_prefetch_b8)
      __builtin_prefetch(W + (size_t)n0 * ldw + (k0 + 32), 0, 1);
  }

  // epilogue: D row = t0 + mf*16 + 8*(lane>=16) + vgpr, col = lane&15
  const int half  = lane >> 4;
  const int nlane = lane & 15;
  float bv[2];
#pragma unroll
  for (int nf = 0; nf < 2; ++nf) bv[nf] = bias[n0 + nf * 16 + nlane];

#pragma unroll
  for (int mf = 0; mf < 4; ++mf)
#pragma unroll
    for (int nf = 0; nf < 2; ++nf) {
      const int col = n0 + nf * 16 + nlane;
#pragma unroll
      for (int e = 0; e < 8; ++e) {
        const int row = t0 + mf * 16 + half * 8 + e;
        float x = acc[mf][nf][e] + bv[nf];
        x = (x > 0.f) ? x : 0.01f * x;  // leaky relu
        Hout[(size_t)row * NP + col] = (bf16)x;
      }
    }
}

// ---------------- fused head: out = tanh(H @ W4^T + b4 + conv_e + conv_i) ---
// 8 waves/block, each wave one 16x32 tile. Dense K=512 pass plus two windowed
// K=224 WMMA passes (the synaptic convolutions) into the same accumulators.
__global__ __launch_bounds__(256) void final_head(
    const bf16* __restrict__ H,
    const bf16* __restrict__ W4b, const float* __restrict__ b4p,
    const bf16* __restrict__ vpe, const bf16* __restrict__ vpi,
    const bf16* __restrict__ wke, const bf16* __restrict__ wki,
    float* __restrict__ out) {
  const int wid  = threadIdx.x >> 5;
  const int lane = threadIdx.x & 31;
  const int t0 = blockIdx.x * BM + wid * 16;

  v8f acc[2];
#pragma unroll
  for (int j = 0; j < 2; ++j)
#pragma unroll
    for (int e = 0; e < 8; ++e) acc[j][e] = 0.f;

  // dense part: H[t, :] @ W4^T
  for (int k0 = 0; k0 < NP; k0 += 32) {
    v16bf a = frag_a_vec(H + (size_t)t0 * NP + k0, NP);
#pragma unroll
    for (int nf = 0; nf < 2; ++nf) {
      v16bf b = frag_b(W4b + (size_t)(nf * 16) * NP + k0, NP);
      acc[nf] = WMMA_BF16(a, b, acc[nf]);
    }
  }
  // synaptic convolutions as windowed GEMMs with flipped kernels
  for (int k0 = 0; k0 < KCONV; k0 += 32) {
    v16bf ae = frag_a_win(vpe + t0 + k0);
    v16bf ai = frag_a_win(vpi + t0 + k0);
#pragma unroll
    for (int nf = 0; nf < 2; ++nf) {
      v16bf be = frag_b(wke + (size_t)(nf * 16) * KCONV + k0, KCONV);
      v16bf bi = frag_b(wki + (size_t)(nf * 16) * KCONV + k0, KCONV);
      acc[nf] = WMMA_BF16(ae, be, acc[nf]);
      acc[nf] = WMMA_BF16(ai, bi, acc[nf]);
    }
  }

  const int half  = lane >> 4;
  const int nlane = lane & 15;
#pragma unroll
  for (int nf = 0; nf < 2; ++nf) {
    const int col = nf * 16 + nlane;
    const float bv = b4p[col];
#pragma unroll
    for (int e = 0; e < 8; ++e) {
      const int row = t0 + half * 8 + e;
      if (row < T_DATA_C && col < SUB_NO_C)
        out[(size_t)row * SUB_NO_C + col] = tanhf(acc[nf][e] + bv);
    }
  }
}

// ---------------- memory-bound row reductions (wave per row) ----------------
__global__ __launch_bounds__(256) void rowsum(const float* __restrict__ S,
                                              int rows, int cols,
                                              float* __restrict__ outv) {
  const int wid = threadIdx.x >> 5, lane = threadIdx.x & 31;
  const int row = blockIdx.x * 8 + wid;
  if (row >= rows) return;
  const float4* p = (const float4*)(S + (size_t)row * cols);
  const int n4 = cols >> 2;
  float s = 0.f;
  for (int i = lane; i < n4; i += 32) {
    float4 v = p[i];
    s += v.x + v.y + v.z + v.w;
  }
#pragma unroll
  for (int off = 16; off; off >>= 1) s += __shfl_xor(s, off, 32);
  if (lane == 0) outv[row] = s;
}

// ---------------- analytic synaptic kernels, pre-flipped, bf16, padded ------
// wk[s][q] = kern[s][T_NO-1-q] for q<200, 0 beyond; rows 20..31 zero.
__global__ void make_kern_bf16(const float* __restrict__ W_syn,
                               const float* __restrict__ Tau,
                               const float* __restrict__ Delta,
                               bf16* __restrict__ wke, bf16* __restrict__ wki) {
  const int i = blockIdx.x * blockDim.x + threadIdx.x;
  if (i >= NCONV * KCONV) return;
  const int s = i / KCONV, q = i % KCONV;
  float ev = 0.f, iv = 0.f;
  if (s < SUB_NO_C && q < T_NO_C) {
    const int d = T_NO_C - 1 - q;  // flip
    const float te = fmaxf((float)d - Delta[s * 2 + 0], 0.f);
    const float ti = fmaxf((float)d - Delta[s * 2 + 1], 0.f);
    const float taue = Tau[s * 2 + 0], taui = Tau[s * 2 + 1];
    const float tte = te / (taue * taue), tti = ti / (taui * taui);
    const float we = W_syn[s * 2 + 0], wi = W_syn[s * 2 + 1];
    ev = tte * expf(-tte) * we * we;
    iv = -(tti * expf(-tti)) * wi * wi;
  }
  wke[i] = (bf16)ev;
  wki[i] = (bf16)iv;
}

// ---------------- fp32 -> bf16 padded weight / bias / window prep -----------
__global__ void cvt_w(const float* __restrict__ W, int rows, int cols,
                      bf16* __restrict__ Wb, int pcols, int total) {
  const int i = blockIdx.x * blockDim.x + threadIdx.x;
  if (i >= total) return;
  const int r = i / pcols, c = i % pcols;
  const float v = (r < rows && c < cols) ? W[(size_t)r * cols + c] : 0.f;
  Wb[i] = (bf16)v;
}

__global__ void pad_vec(const float* __restrict__ b, int n,
                        float* __restrict__ bp, int np) {
  const int i = blockIdx.x * blockDim.x + threadIdx.x;
  if (i >= np) return;
  bp[i] = (i < n) ? b[i] : 0.f;
}

// vp[i] = src[i - 199] (bf16), zero outside [0, n)
__global__ void build_vpad_f(const float* __restrict__ src,
                             bf16* __restrict__ vp, int len, int n) {
  const int i = blockIdx.x * blockDim.x + threadIdx.x;
  if (i >= len) return;
  const int j = i - (T_NO_C - 1);
  vp[i] = (bf16)((j >= 0 && j < n) ? src[j] : 0.f);
}

// ---------------------------------------------------------------------------
extern "C" void kernel_launch(void* const* d_in, const int* in_sizes, int n_in,
                              void* d_out, int out_size, void* d_ws,
                              size_t ws_size, hipStream_t stream) {
  const float* V     = (const float*)d_in[0];
  const float* S_e   = (const float*)d_in[1];
  const float* S_i   = (const float*)d_in[2];
  const float* W1    = (const float*)d_in[3];
  const float* b1    = (const float*)d_in[4];
  const float* W2    = (const float*)d_in[5];
  const float* b2    = (const float*)d_in[6];
  const float* W3    = (const float*)d_in[7];
  const float* b3    = (const float*)d_in[8];
  const float* W4    = (const float*)d_in[9];
  const float* b4    = (const float*)d_in[10];
  const float* W_syn = (const float*)d_in[11];
  const float* Tau   = (const float*)d_in[12];
  const float* Delta = (const float*)d_in[13];
  float* out = (float*)d_out;

  char* wsp = (char*)d_ws;
  auto alloc = [&](size_t bytes) -> char* {
    char* p = wsp;
    wsp += (bytes + 255) & ~(size_t)255;
    return p;
  };
  float* syn_e = (float*)alloc((size_t)T_DATA_C * 4);
  float* syn_i = (float*)alloc((size_t)T_DATA_C * 4);
  bf16*  vpadV = (bf16*)alloc((size_t)VPAD_LEN * 2);
  bf16*  vpe   = (bf16*)alloc((size_t)VPAD_SYN * 2);
  bf16*  vpi   = (bf16*)alloc((size_t)VPAD_SYN * 2);
  bf16*  wke   = (bf16*)alloc((size_t)NCONV * KCONV * 2);
  bf16*  wki   = (bf16*)alloc((size_t)NCONV * KCONV * 2);
  bf16*  W1b   = (bf16*)alloc((size_t)NP * K1P * 2);
  bf16*  W2b   = (bf16*)alloc((size_t)NP * NP * 2);
  bf16*  W3b   = (bf16*)alloc((size_t)NP * NP * 2);
  bf16*  W4b   = (bf16*)alloc((size_t)NCONV * NP * 2);
  float* b1p   = (float*)alloc((size_t)NP * 4);
  float* b2p   = (float*)alloc((size_t)NP * 4);
  float* b3p   = (float*)alloc((size_t)NP * 4);
  float* b4p   = (float*)alloc((size_t)NCONV * 4);
  bf16*  H1    = (bf16*)alloc((size_t)MROWS * NP * 2);
  bf16*  H2    = (bf16*)alloc((size_t)MROWS * NP * 2);

  // 1) memory-bound row sums (the 500 MB streaming floor)
  rowsum<<<dim3((T_DATA_C + 7) / 8), dim3(256), 0, stream>>>(S_e, T_DATA_C, 2000, syn_e);
  rowsum<<<dim3((T_DATA_C + 7) / 8), dim3(256), 0, stream>>>(S_i, T_DATA_C, 500, syn_i);

  // 2) bf16 prep: padded windows, flipped conv kernels, weights, biases
  build_vpad_f<<<dim3((VPAD_LEN + 255) / 256), dim3(256), 0, stream>>>(V, vpadV, VPAD_LEN, T_DATA_C);
  build_vpad_f<<<dim3((VPAD_SYN + 255) / 256), dim3(256), 0, stream>>>(syn_e, vpe, VPAD_SYN, T_DATA_C);
  build_vpad_f<<<dim3((VPAD_SYN + 255) / 256), dim3(256), 0, stream>>>(syn_i, vpi, VPAD_SYN, T_DATA_C);
  make_kern_bf16<<<dim3((NCONV * KCONV + 255) / 256), dim3(256), 0, stream>>>(W_syn, Tau, Delta, wke, wki);
  cvt_w<<<dim3((NP * K1P + 255) / 256), dim3(256), 0, stream>>>(W1, HID_C, WIN_C, W1b, K1P, NP * K1P);
  cvt_w<<<dim3((NP * NP + 255) / 256), dim3(256), 0, stream>>>(W2, HID_C, HID_C, W2b, NP, NP * NP);
  cvt_w<<<dim3((NP * NP + 255) / 256), dim3(256), 0, stream>>>(W3, HID_C, HID_C, W3b, NP, NP * NP);
  cvt_w<<<dim3((NCONV * NP + 255) / 256), dim3(256), 0, stream>>>(W4, SUB_NO_C, HID_C, W4b, NP, NCONV * NP);
  pad_vec<<<dim3((NP + 255) / 256), dim3(256), 0, stream>>>(b1, HID_C, b1p, NP);
  pad_vec<<<dim3((NP + 255) / 256), dim3(256), 0, stream>>>(b2, HID_C, b2p, NP);
  pad_vec<<<dim3((NP + 255) / 256), dim3(256), 0, stream>>>(b3, HID_C, b3p, NP);
  pad_vec<<<dim3((NCONV + 255) / 256), dim3(256), 0, stream>>>(b4, SUB_NO_C, b4p, NCONV);

  // 3) WMMA MLP: windowed layer 1, two hidden layers, fused head (+conv+tanh)
  dim3 g(MBLK, NP / BN);  // 391 x 4
  mlp_gemm<true ><<<g, dim3(256), 0, stream>>>(vpadV, 1, W1b, K1P, b1p, K1P, H1);
  mlp_gemm<false><<<g, dim3(256), 0, stream>>>(H1, NP, W2b, NP, b2p, NP, H2);
  mlp_gemm<false><<<g, dim3(256), 0, stream>>>(H2, NP, W3b, NP, b3p, NP, H1);
  final_head<<<dim3(MBLK), dim3(256), 0, stream>>>(H1, W4b, b4p, vpe, vpi, wke, wki, out);
}
